// YoloV3HeadTest_33414845563587
// MI455X (gfx1250) — matrix operand order repaired and checked by reading, not compiled, not run on hardware
//
#include <hip/hip_runtime.h>
#include <hip/hip_bf16.h>
#include <math.h>

// ---------------------------------------------------------------------------
// CDNA5 (gfx1250) YOLOv3 head. Only batch element 0 reaches the output
// (reference decodes pm[0]), so all compute is batch-0 only.
// Conv3x3 = implicit GEMM with v_wmma_f32_16x16x32_bf16 (wave32).
// B (weight) tiles are staged into LDS by the Tensor Data Mover
// (tensor_load_to_lds, TENSORcnt), with hardware LDS padding giving a
// conflict-free 34-element row stride.
// ---------------------------------------------------------------------------

typedef __bf16 v16bf __attribute__((ext_vector_type(16)));
typedef float  v8f   __attribute__((ext_vector_type(8)));
typedef unsigned uint4v __attribute__((ext_vector_type(4)));
typedef unsigned uint8v __attribute__((ext_vector_type(8)));

union FragAB { v16bf v; unsigned u[8]; };

__device__ __forceinline__ unsigned short f2bf(float f) {
    unsigned u = __float_as_uint(f);
    u += 0x7fffu + ((u >> 16) & 1u);           // round-to-nearest-even
    return (unsigned short)(u >> 16);
}
__device__ __forceinline__ float bf2f(unsigned short h) {
    return __uint_as_float(((unsigned)h) << 16);
}
__device__ __forceinline__ float sigm(float x) {
    return 1.0f / (1.0f + expf(-x));
}

// ---------------------------------------------------------------------------
// Repack conv weights wb[O][C][3][3] (f32) -> bf16, tile-contiguous:
//   Bt2[ ((r*(CIN/32) + cb)*COUT + n)*32 + kc ] = wb[n][cb*32+kc][r]
// so each GEMM K-step's 128(n) x 32(k) B tile is one contiguous 8KB run
// (ready for a single tensor_load_to_lds DMA).
// ---------------------------------------------------------------------------
template<int CIN, int COUT>
__global__ void repack_w(const float* __restrict__ wb,
                         unsigned short* __restrict__ bt) {
    const int total = 9 * CIN * COUT;
    for (int idx = blockIdx.x * blockDim.x + threadIdx.x; idx < total;
         idx += gridDim.x * blockDim.x) {
        int kc = idx & 31;
        int n  = (idx >> 5) % COUT;
        int cb = (idx / (32 * COUT)) % (CIN / 32);
        int r  = idx / (32 * COUT * (CIN / 32));
        bt[idx] = f2bf(wb[(n * CIN + cb * 32 + kc) * 9 + r]);
    }
}

// ---------------------------------------------------------------------------
// Implicit-GEMM 3x3 conv (SAME), batch 0 only, fused BN-fold + LeakyReLU(0.1).
// Block = 256 threads = 8 waves; block tile 64(M) x 128(N);
// each wave owns a 32x32 block = 2x2 WMMA 16x16 tiles. K-step = 32 channels
// of one tap r.
// ---------------------------------------------------------------------------
template<int CIN, int COUT, int H, int W>
__global__ __launch_bounds__(256) void conv3x3_wmma(
    const float* __restrict__ feat,          // [8,CIN,H,W]; batch 0 read
    const unsigned short* __restrict__ bt,   // bf16 tile-contiguous (see repack)
    const float* __restrict__ bns, const float* __restrict__ bnb,
    unsigned short* __restrict__ ybf)        // bf16 [H*W][COUT]
{
    constexpr int HW  = H * W;
    constexpr int LDA = 34;                  // padded LDS row stride (bf16 elems)
    __shared__ __align__(16) unsigned short Ash[64  * LDA]; // A: [m][k]
    __shared__ __align__(16) unsigned short Bsh[128 * LDA]; // B: [n][k] (TDM-filled)

    const int tid  = threadIdx.x;
    const int lane = tid & 31;
    const int wid  = tid >> 5;
    const int mi   = wid & 1;                // which 32-row block of the tile
    const int ni   = wid >> 1;               // which 32-col block of the tile
    const int blockM = blockIdx.x * 64;
    const int blockN = blockIdx.y * 128;

    v8f acc[2][2];
    #pragma unroll
    for (int a = 0; a < 2; ++a)
        #pragma unroll
        for (int b = 0; b < 2; ++b)
            #pragma unroll
            for (int j = 0; j < 8; ++j) acc[a][b][j] = 0.0f;

    // ---- Tensor DMA descriptor (uniform; only the global address varies) ----
    // D# group1: data_size=2B, pad_enable=1, pad_interval=3 (every 16 DWORDs),
    // pad_amount=0 (1 DWORD) -> LDS row stride 34 elems. 1-D tile of 4096 elems.
    const unsigned ldsB = (unsigned)(unsigned long long)(uintptr_t)&Bsh[0];
    uint8v g1;
    g1[0] = 0x00010000u | (1u << 20) | (3u << 22); // data_size=1(2B), pad en/int
    g1[1] = 4096u << 16;                           // tensor_dim0[15:0]=4096
    g1[2] = 1u << 16;                              // tensor_dim1=1
    g1[3] = 4096u << 16;                           // tile_dim0=4096
    g1[4] = 0u;                                    // tile_dim1/2 unused
    g1[5] = 4096u;                                 // tensor_dim0_stride lo
    g1[6] = 0u;
    g1[7] = 0u;

    // cooperative A-fill index precompute
    const int am  = tid & 63;                // A: m within tile (coalesced dim)
    const int akb = (tid >> 6) << 3;         // A: k base (0,8,16,24)
    const int pos = blockM + am;
    const int hh  = pos / W, ww = pos % W;

    for (int r = 0; r < 9; ++r) {
        const int dy = r / 3 - 1, dx = r % 3 - 1;
        const int hs = hh + dy, wcol = ww + dx;
        const bool inb = (hs >= 0) && (hs < H) && (wcol >= 0) && (wcol < W);
        const int sp = inb ? (hs * W + wcol) : 0;   // safe fallback address
        const float* fA = feat + sp;

        for (int cb = 0; cb < CIN / 32; ++cb) {
            __syncthreads();                 // LDS free from previous step

            // ---- wave 0: launch B-tile DMA (8KB contiguous -> padded LDS) ----
            if (wid == 0) {
                const unsigned long long ga = (unsigned long long)(uintptr_t)
                    (bt + ((size_t)(r * (CIN / 32) + cb) * COUT + blockN) * 32);
                uint4v g0;
                g0[0] = 1u;                              // count=1 (valid D#)
                g0[1] = ldsB;                            // lds_addr
                g0[2] = (unsigned)ga;                    // global_addr[31:0]
                g0[3] = ((unsigned)(ga >> 32) & 0x01ffffffu) | (2u << 30); // type=2
                asm volatile("tensor_load_to_lds %0, %1"
                             :: "s"(g0), "s"(g1) : "memory");
            }

            // ---- all threads: stage A tile (branchless halo handling) ----
            #pragma unroll
            for (int j = 0; j < 8; ++j) {
                int k = akb + j;
                float raw = fA[(size_t)(cb * 32 + k) * HW];
                float v = inb ? raw : 0.0f;
                Ash[am * LDA + k] = f2bf(v);
            }

            if (wid == 0) __builtin_amdgcn_s_wait_tensorcnt(0);
            __syncthreads();                 // A + B tiles visible to all waves

            // ---- fragment loads per documented CDNA5 WMMA VGPR layouts ----
            FragAB afr[2], bfr[2];
            #pragma unroll
            for (int t = 0; t < 2; ++t) {
                const int row = mi * 32 + t * 16 + (lane & 15);
                #pragma unroll
                for (int j = 0; j < 8; ++j) {
                    // A 16x32 bf16: K = (j>=4?16:0) + laneHalf*8 + (j&3)*2
                    int ke = ((j & 4) << 2) + ((lane >> 4) << 3) + ((j & 3) << 1);
                    afr[t].u[j] = *(const unsigned*)(&Ash[row * LDA + ke]);
                }
                const int col = ni * 32 + t * 16 + (lane & 15);
                #pragma unroll
                for (int j = 0; j < 8; ++j) {
                    // B 32x16 bf16: K = (j>=4?16:0) + laneHalf*8 + (j&3)*2
                    int ke = ((j & 4) << 2) + ((lane >> 4) << 3) + ((j & 3) << 1);
                    bfr[t].u[j] = *(const unsigned*)(&Bsh[col * LDA + ke]);
                }
            }
            #pragma unroll
            for (int a = 0; a < 2; ++a)
                #pragma unroll
                for (int b = 0; b < 2; ++b)
                    acc[a][b] = __builtin_amdgcn_wmma_f32_16x16x32_bf16(
                        false, afr[a].v, false, bfr[b].v,
                        (short)0, acc[a][b], false, false);
        }
    }

    // ---- epilogue: BN fold + LeakyReLU(0.1), store bf16 y[m][n] ----
    #pragma unroll
    for (int a = 0; a < 2; ++a) {
        #pragma unroll
        for (int b = 0; b < 2; ++b) {
            const int n = blockN + ni * 32 + b * 16 + (lane & 15);
            const float sc = bns[n], bi = bnb[n];
            #pragma unroll
            for (int j = 0; j < 8; ++j) {
                const int m = blockM + mi * 32 + a * 16 + ((lane >> 4) << 3) + j;
                float v = acc[a][b][j] * sc + bi;
                v = v > 0.0f ? v : 0.1f * v;
                ybf[(size_t)m * COUT + n] = f2bf(v);
            }
        }
    }
}

// ---------------------------------------------------------------------------
// 1x1 conv (27 channels) + YOLO decode, fused. One thread per spatial pos.
// Output layout: bbox[64512][4] | cls[64512][4] | conf[64512].
// ---------------------------------------------------------------------------
template<int COUT, int W>
__global__ __launch_bounds__(64) void head_decode(
    const unsigned short* __restrict__ ybf, const float* __restrict__ wp,
    const float* __restrict__ bp, float* __restrict__ out,
    int rowOff, float s,
    float bw0, float bh0, float bw1, float bh1, float bw2, float bh2)
{
    constexpr int NT = 64512;
    __shared__ float wpch[27 * 64];
    const int tid = threadIdx.x;
    const int m = blockIdx.x * 64 + tid;

    float acc[27];
    #pragma unroll
    for (int q = 0; q < 27; ++q) acc[q] = bp[q];

    for (int k0 = 0; k0 < COUT; k0 += 64) {
        __syncthreads();
        #pragma unroll
        for (int q = 0; q < 27; ++q)
            wpch[q * 64 + tid] = wp[q * COUT + k0 + tid];
        __syncthreads();
        const unsigned short* yrow = ybf + (size_t)m * COUT + k0;
        #pragma unroll 4
        for (int j = 0; j < 64; ++j) {
            const float yv = bf2f(yrow[j]);
            #pragma unroll
            for (int q = 0; q < 27; ++q) acc[q] += yv * wpch[q * 64 + j];
        }
    }

    const float bwv[3] = {bw0, bw1, bw2};
    const float bhv[3] = {bh0, bh1, bh2};
    const float cx = (float)(m % W) * s + 0.5f * s;
    const float cy = (float)(m / W) * s + 0.5f * s;
    #pragma unroll
    for (int a = 0; a < 3; ++a) {
        const int row = rowOff + m * 3 + a;
        const float xc = (sigm(acc[a * 9 + 0]) - 0.5f) * s + cx;
        const float yc = (sigm(acc[a * 9 + 1]) - 0.5f) * s + cy;
        const float bw = expf(acc[a * 9 + 2]) * bwv[a];
        const float bh = expf(acc[a * 9 + 3]) * bhv[a];
        out[row * 4 + 0] = xc - 0.5f * bw;
        out[row * 4 + 1] = yc - 0.5f * bh;
        out[row * 4 + 2] = xc + 0.5f * bw;
        out[row * 4 + 3] = yc + 0.5f * bh;
        #pragma unroll
        for (int j = 0; j < 4; ++j)
            out[4 * NT + row * 4 + j] = sigm(acc[a * 9 + 5 + j]);
        out[8 * NT + row] = sigm(acc[a * 9 + 4]);
    }
}

// ---------------------------------------------------------------------------
extern "C" void kernel_launch(void* const* d_in, const int* in_sizes, int n_in,
                              void* d_out, int out_size, void* d_ws, size_t ws_size,
                              hipStream_t stream) {
    (void)in_sizes; (void)n_in; (void)out_size; (void)ws_size;
    float* out = (float*)d_out;
    // workspace: [0, 9.4MB) = repacked bf16 weights; [9.4MB, +8MB) = bf16 y
    unsigned short* bt  = (unsigned short*)d_ws;
    unsigned short* ybf = (unsigned short*)((char*)d_ws +
                           (size_t)9 * 512 * 1024 * sizeof(unsigned short));

    // ---------------- level 0: 512 -> 1024, 32x32, stride 32 ----------------
    {
        const float* feat = (const float*)d_in[0];
        const float* wb   = (const float*)d_in[1];
        const float* bns  = (const float*)d_in[2];
        const float* bnb  = (const float*)d_in[3];
        const float* wp   = (const float*)d_in[4];
        const float* bp   = (const float*)d_in[5];
        repack_w<512, 1024><<<dim3(4096), dim3(256), 0, stream>>>(wb, bt);
        conv3x3_wmma<512, 1024, 32, 32>
            <<<dim3(16, 8), dim3(256), 0, stream>>>(feat, bt, bns, bnb, ybf);
        head_decode<1024, 32><<<dim3(16), dim3(64), 0, stream>>>(
            ybf, wp, bp, out, /*rowOff=*/0, /*s=*/32.0f,
            116.f, 90.f, 156.f, 198.f, 373.f, 326.f);
    }
    // ---------------- level 1: 256 -> 512, 64x64, stride 16 -----------------
    {
        const float* feat = (const float*)d_in[6];
        const float* wb   = (const float*)d_in[7];
        const float* bns  = (const float*)d_in[8];
        const float* bnb  = (const float*)d_in[9];
        const float* wp   = (const float*)d_in[10];
        const float* bp   = (const float*)d_in[11];
        repack_w<256, 512><<<dim3(2048), dim3(256), 0, stream>>>(wb, bt);
        conv3x3_wmma<256, 512, 64, 64>
            <<<dim3(64, 4), dim3(256), 0, stream>>>(feat, bt, bns, bnb, ybf);
        head_decode<512, 64><<<dim3(64), dim3(64), 0, stream>>>(
            ybf, wp, bp, out, /*rowOff=*/3072, /*s=*/16.0f,
            30.f, 61.f, 62.f, 45.f, 59.f, 119.f);
    }
    // ---------------- level 2: 128 -> 256, 128x128, stride 8 ----------------
    {
        const float* feat = (const float*)d_in[12];
        const float* wb   = (const float*)d_in[13];
        const float* bns  = (const float*)d_in[14];
        const float* bnb  = (const float*)d_in[15];
        const float* wp   = (const float*)d_in[16];
        const float* bp   = (const float*)d_in[17];
        repack_w<128, 256><<<dim3(1024), dim3(256), 0, stream>>>(wb, bt);
        conv3x3_wmma<128, 256, 128, 128>
            <<<dim3(256, 2), dim3(256), 0, stream>>>(feat, bt, bns, bnb, ybf);
        head_decode<256, 128><<<dim3(256), dim3(64), 0, stream>>>(
            ybf, wp, bp, out, /*rowOff=*/15360, /*s=*/8.0f,
            10.f, 13.f, 16.f, 30.f, 33.f, 23.f);
    }
}
